// ComplexMultiHeadAttention_58308476010503
// MI455X (gfx1250) — matrix operand order repaired and checked by reading, not compiled, not run on hardware
//
#include <hip/hip_runtime.h>
#include <hip/hip_bf16.h>

typedef __attribute__((ext_vector_type(16))) __bf16 v16bf;
typedef __attribute__((ext_vector_type(8)))  __bf16 v8bf;
typedef __attribute__((ext_vector_type(8)))  float  v8f;
typedef __attribute__((ext_vector_type(4)))  unsigned int tdm_u32x4;
typedef __attribute__((ext_vector_type(8)))  int          tdm_i32x8;
typedef __attribute__((ext_vector_type(4)))  int          tdm_i32x4;

#define D_MODEL 768
#define SEQ     2048
#define NBATCH  2
#define NH      12
#define DH      64
#define MTOT    (NBATCH*SEQ)   // 4096

#if __has_builtin(__builtin_amdgcn_tensor_load_to_lds)
#define HAVE_TDM 1
#else
#define HAVE_TDM 0
#endif

// ---------------------------------------------------------------------------
// Fragment loaders (wave32 WMMA; A/B use this code's contiguous k order, C/D as in the ISA)
// A 16x32 bf16: lane L -> row M=L%16; half=L/16; elem e -> K=16*half+e.
//               Two aligned 16B chunks.
// B 32x16 bf16: lane L -> col N=L%16; elem e -> K=16*half+e. One 32B run.
// C/D 16x16 f32: VGPR r, lane L -> (M = r+8*half, N = L%16).
// ---------------------------------------------------------------------------
__device__ inline v16bf load_afrag(const __bf16* __restrict__ base, int ld,
                                   int row0, int kb, int lane) {
  const int m = row0 + (lane & 15);
  const int half = lane >> 4;
  const v8bf* p0 = reinterpret_cast<const v8bf*>(base + (size_t)m * ld + kb + 16 * half);
  const v8bf* p1 = reinterpret_cast<const v8bf*>(base + (size_t)m * ld + kb + 16 * half + 8);
  v8bf x0 = *p0, x1 = *p1;
  v16bf a;
#pragma unroll
  for (int e = 0; e < 8; ++e) { a[e] = x0[e]; a[8 + e] = x1[e]; }
  return a;
}

__device__ inline v16bf load_bfrag(const __bf16* __restrict__ base, int ld,
                                   int n0, int kb, int lane) {
  const int n = n0 + (lane & 15);
  const int half = lane >> 4;
  const v8bf* p0 = reinterpret_cast<const v8bf*>(base + (size_t)n * ld + kb + 16 * half);
  v8bf x0 = p0[0], x1 = p0[1];
  v16bf b;
#pragma unroll
  for (int e = 0; e < 8; ++e) { b[e] = x0[e]; b[8 + e] = x1[e]; }
  return b;
}

__device__ inline v8f wmma_bf16(v16bf a, v16bf b, v8f c) {
  return __builtin_amdgcn_wmma_f32_16x16x32_bf16(
      false, a, false, b, (short)0, c, false, false);
}

// ---------------------------------------------------------------------------
// fp32 -> bf16 bulk convert (8 elems/thread, vector loads/stores)
// ---------------------------------------------------------------------------
__global__ __launch_bounds__(256)
void cvt_f32_bf16_kernel(const float* __restrict__ src, __bf16* __restrict__ dst, int n) {
  int i = (blockIdx.x * 256 + threadIdx.x) * 8;
  if (i + 8 > n) return;
  float4 a = *reinterpret_cast<const float4*>(src + i);
  float4 b = *reinterpret_cast<const float4*>(src + i + 4);
  v8bf o;
  o[0] = (__bf16)a.x; o[1] = (__bf16)a.y; o[2] = (__bf16)a.z; o[3] = (__bf16)a.w;
  o[4] = (__bf16)b.x; o[5] = (__bf16)b.y; o[6] = (__bf16)b.z; o[7] = (__bf16)b.w;
  *reinterpret_cast<v8bf*>(dst + i) = o;
}

// ---------------------------------------------------------------------------
// TDM: DMA a [rows=128 x cols=32] bf16 tile (tensor row stride 768 elems) into
// LDS with 16B padding after each 64B row -> padded LDS stride of 40 bf16.
// D# layout per CDNA5 ISA 8.3/8.4.
// ---------------------------------------------------------------------------
#define BM 128
#define BN 128
#define BK 32
#define LDT 40   // BK + 8 pad (80B LDS row stride)

#if HAVE_TDM
__device__ inline void tdm_load_tile_128x32(unsigned lds_off,
                                            const __bf16* gptr,
                                            unsigned tensor_rows) {
  unsigned long long ga = (unsigned long long)(uintptr_t)gptr;
  tdm_u32x4 g0;
  g0[0] = 1u;                                            // count=1, no gather
  g0[1] = lds_off;                                       // lds_addr (bytes)
  g0[2] = (unsigned)(ga & 0xffffffffu);                  // global_addr lo
  g0[3] = (unsigned)((ga >> 32) & 0x1ffffffu) | (2u << 30);  // addr hi | type=2
  tdm_i32x8 g1;
  // data_size=1(2B) | pad_enable | pad_interval=3 (16 dwords) | pad_amount=3 (4 dwords)
  g1[0] = (int)((1u << 16) | (1u << 20) | (3u << 22) | (3u << 25));
  g1[1] = (int)((unsigned)(D_MODEL & 0xffff) << 16);     // tensor_dim0 lo16
  g1[2] = (int)(((unsigned)D_MODEL >> 16) | ((tensor_rows & 0xffffu) << 16)); // dim0 hi | dim1 lo
  g1[3] = (int)((tensor_rows >> 16) | (32u << 16));      // dim1 hi | tile_dim0=32
  g1[4] = (int)128;                                      // tile_dim1=128, tile_dim2=0
  g1[5] = (int)D_MODEL;                                  // tensor_dim0_stride lo32
  g1[6] = 0;                                             // stride hi | dim1_stride lo
  g1[7] = 0;
  tdm_i32x4 z4 = {0, 0, 0, 0};
#if defined(__clang_major__) && (__clang_major__ >= 23)
  tdm_i32x8 z8 = {0, 0, 0, 0, 0, 0, 0, 0};
  __builtin_amdgcn_tensor_load_to_lds(g0, g1, z4, z4, z8, 0);
#else
  __builtin_amdgcn_tensor_load_to_lds(g0, g1, z4, z4, 0);
#endif
}
#endif

// ---------------------------------------------------------------------------
// Projection GEMM:  out_bf16[m][n] = sum_k xbf[m][k] * Wbf[n][k]   (x @ W^T)
// 128x128 block tile, BK=32, 4 waves each computing 64x64 (16 WMMAs/step).
// z==4 (the V projection) is stored transposed: vt[b][h][dh][s].
// ---------------------------------------------------------------------------
__device__ inline void tile_mma(const __bf16* Asb, const __bf16* Bsb,
                                int wm, int wn, int lane, v8f acc[4][4]) {
  v16bf af[4], bfr[4];
#pragma unroll
  for (int i = 0; i < 4; ++i) af[i]  = load_afrag(Asb, LDT, wm + i * 16, 0, lane);
#pragma unroll
  for (int j = 0; j < 4; ++j) bfr[j] = load_bfrag(Bsb, LDT, wn + j * 16, 0, lane);
#pragma unroll
  for (int i = 0; i < 4; ++i)
#pragma unroll
    for (int j = 0; j < 4; ++j)
      acc[i][j] = wmma_bf16(af[i], bfr[j], acc[i][j]);
}

__global__ __launch_bounds__(128)
void proj_gemm_kernel(const __bf16* __restrict__ xbf, const __bf16* __restrict__ wbf,
                      __bf16* __restrict__ proj, __bf16* __restrict__ vt) {
  __shared__ __align__(16) __bf16 As[2][BM * LDT];
  __shared__ __align__(16) __bf16 Bs[2][BM * LDT];

  const int z = blockIdx.z;
  const __bf16* __restrict__ Wz = wbf + (size_t)z * D_MODEL * D_MODEL;

  const int tid  = threadIdx.x;
  const int lane = tid & 31;
  const int w    = tid >> 5;
  const int wm   = (w >> 1) * 64;
  const int wn   = (w & 1) * 64;
  const int m0   = blockIdx.x * BM;
  const int n0   = blockIdx.y * BN;

  const __bf16* xrow = xbf + (size_t)m0 * D_MODEL;
  const __bf16* wrow = Wz  + (size_t)n0 * D_MODEL;

  v8f acc[4][4];
#pragma unroll
  for (int i = 0; i < 4; ++i)
#pragma unroll
    for (int j = 0; j < 4; ++j)
#pragma unroll
      for (int e = 0; e < 8; ++e) acc[i][j][e] = 0.0f;

  const int nk = D_MODEL / BK;

#if HAVE_TDM
  if (tid < 32) {  // wave 0 drives the DMA pipeline
    tdm_load_tile_128x32((unsigned)(uintptr_t)(void*)&As[0][0], xrow, (unsigned)MTOT);
    tdm_load_tile_128x32((unsigned)(uintptr_t)(void*)&Bs[0][0], wrow, (unsigned)D_MODEL);
  }
  for (int kt = 0; kt < nk; ++kt) {
    const int cur = kt & 1, nxt = cur ^ 1;
    if (tid < 32) {
      if (kt + 1 < nk) {
        tdm_load_tile_128x32((unsigned)(uintptr_t)(void*)&As[nxt][0],
                             xrow + (kt + 1) * BK, (unsigned)MTOT);
        tdm_load_tile_128x32((unsigned)(uintptr_t)(void*)&Bs[nxt][0],
                             wrow + (kt + 1) * BK, (unsigned)D_MODEL);
        __builtin_amdgcn_s_wait_tensorcnt(2);   // current pair complete (in-order)
      } else {
        __builtin_amdgcn_s_wait_tensorcnt(0);
      }
    }
    __syncthreads();                            // publish current tiles
    tile_mma(&As[cur][0], &Bs[cur][0], wm, wn, lane, acc);
    __syncthreads();                            // all readers done before reuse
  }
#else
  for (int kt = 0; kt < nk; ++kt) {
    const int cur = kt & 1;
    __syncthreads();
    {
      const __bf16* xr = xrow + (size_t)tid * D_MODEL + kt * BK;
      const __bf16* wr = wrow + (size_t)tid * D_MODEL + kt * BK;
      v8bf* ad = reinterpret_cast<v8bf*>(&As[cur][tid * LDT]);
      v8bf* bd = reinterpret_cast<v8bf*>(&Bs[cur][tid * LDT]);
#pragma unroll
      for (int c = 0; c < 4; ++c) {
        ad[c] = *reinterpret_cast<const v8bf*>(xr + c * 8);
        bd[c] = *reinterpret_cast<const v8bf*>(wr + c * 8);
      }
    }
    __syncthreads();
    tile_mma(&As[cur][0], &Bs[cur][0], wm, wn, lane, acc);
  }
#endif

  // ---- store D tiles as bf16 (z==4 stored transposed for the attention PV) ----
  const int half = lane >> 4;
  const int l16  = lane & 15;
  if (z == 4) {
#pragma unroll
    for (int i = 0; i < 4; ++i)
#pragma unroll
      for (int j = 0; j < 4; ++j)
#pragma unroll
        for (int r = 0; r < 8; ++r) {
          int m = m0 + wm + i * 16 + r + 8 * half;   // b*SEQ + s
          int n = n0 + wn + j * 16 + l16;            // h*DH + dh
          int bI = m >> 11, s = m & (SEQ - 1);
          int hI = n >> 6,  dh = n & (DH - 1);
          vt[(((size_t)bI * NH + hI) * DH + dh) * SEQ + s] = (__bf16)acc[i][j][r];
        }
  } else {
    __bf16* __restrict__ out = proj + (size_t)z * MTOT * D_MODEL;
#pragma unroll
    for (int i = 0; i < 4; ++i)
#pragma unroll
      for (int j = 0; j < 4; ++j)
#pragma unroll
        for (int r = 0; r < 8; ++r) {
          int m = m0 + wm + i * 16 + r + 8 * half;
          int n = n0 + wn + j * 16 + l16;
          out[(size_t)m * D_MODEL + n] = (__bf16)acc[i][j][r];
        }
  }
}

// ---------------------------------------------------------------------------
// Flash-style complex-magnitude attention. Block = 64 q rows of one (b,h);
// each of 4 waves owns 16 q rows and runs fully independently (no barriers):
// K-frags direct from global, V-frags direct from the transposed vt buffer.
// ---------------------------------------------------------------------------
__global__ __launch_bounds__(128)
void attn_kernel(const __bf16* __restrict__ qr, const __bf16* __restrict__ qi,
                 const __bf16* __restrict__ kr, const __bf16* __restrict__ ki,
                 const __bf16* __restrict__ vt, float* __restrict__ out) {
  __shared__ __align__(16) __bf16 Pt[4 * 16 * 72];  // per-wave P tile (16x64, ld 72)

  const int tid  = threadIdx.x;
  const int lane = tid & 31;
  const int w    = tid >> 5;
  const int half = lane >> 4;
  const int l16  = lane & 15;
  const int q0   = blockIdx.x * 64;
  const int h    = blockIdx.y;
  const int b    = blockIdx.z;

  const size_t rowbase = (size_t)(b * SEQ) * D_MODEL + h * DH;
  const __bf16* qrb = qr + rowbase;
  const __bf16* qib = qi + rowbase;
  const __bf16* krb = kr + rowbase;
  const __bf16* kib = ki + rowbase;
  const __bf16* vtb = vt + ((size_t)(b * NH + h) * DH) * SEQ;  // [dh][s]

  // Q fragments held in registers for the whole block (2 dh-steps of 32)
  v16bf qrf[2], qif[2];
#pragma unroll
  for (int d = 0; d < 2; ++d) {
    qrf[d] = load_afrag(qrb, D_MODEL, q0 + w * 16, d * 32, lane);
    qif[d] = load_afrag(qib, D_MODEL, q0 + w * 16, d * 32, lane);
  }

  float m_i[8], l_i[8];
  v8f O[4];
#pragma unroll
  for (int r = 0; r < 8; ++r) { m_i[r] = -1e30f; l_i[r] = 0.0f; }
#pragma unroll
  for (int n = 0; n < 4; ++n)
#pragma unroll
    for (int r = 0; r < 8; ++r) O[n][r] = 0.0f;

  const float scale = 0.125f;  // 1/sqrt(64)
  __bf16* pw = Pt + w * 16 * 72;

  for (int k0 = 0; k0 < SEQ; k0 += 64) {
    if (k0 + 64 < SEQ) {  // prefetch next K/V tiles into cache
      __builtin_prefetch(krb + (size_t)(k0 + 64) * D_MODEL, 0, 0);
      __builtin_prefetch(kib + (size_t)(k0 + 64) * D_MODEL, 0, 0);
      __builtin_prefetch(vtb + (size_t)(k0 + 64), 0, 0);
    }

    // ---- scores: sr = qr.kr + qi.ki ; si = qi.kr - qr.ki ----
    float p[4][8];
    float mx[8];
#pragma unroll
    for (int r = 0; r < 8; ++r) mx[r] = -1e30f;

#pragma unroll
    for (int j = 0; j < 4; ++j) {
      v8f sr, t1, t2;
#pragma unroll
      for (int e = 0; e < 8; ++e) { sr[e] = 0.0f; t1[e] = 0.0f; t2[e] = 0.0f; }
#pragma unroll
      for (int d = 0; d < 2; ++d) {
        v16bf krf = load_bfrag(krb + (size_t)k0 * D_MODEL, D_MODEL, j * 16, d * 32, lane);
        v16bf kif = load_bfrag(kib + (size_t)k0 * D_MODEL, D_MODEL, j * 16, d * 32, lane);
        sr = wmma_bf16(qrf[d], krf, sr);
        sr = wmma_bf16(qif[d], kif, sr);
        t1 = wmma_bf16(qif[d], krf, t1);
        t2 = wmma_bf16(qrf[d], kif, t2);
      }
#pragma unroll
      for (int r = 0; r < 8; ++r) {
        float a = sr[r] * scale;
        float bb = (t1[r] - t2[r]) * scale;
        float sc = sqrtf(a * a + bb * bb);
        p[j][r] = sc;
        mx[r] = fmaxf(mx[r], sc);
      }
    }

    // ---- online softmax (row stats across the 16 lanes of each EXEC half) ----
#pragma unroll
    for (int r = 0; r < 8; ++r) {
      mx[r] = fmaxf(mx[r], __shfl_xor(mx[r], 8, 32));
      mx[r] = fmaxf(mx[r], __shfl_xor(mx[r], 4, 32));
      mx[r] = fmaxf(mx[r], __shfl_xor(mx[r], 2, 32));
      mx[r] = fmaxf(mx[r], __shfl_xor(mx[r], 1, 32));
    }
    float alpha[8], rs[8];
#pragma unroll
    for (int r = 0; r < 8; ++r) {
      float mnew = fmaxf(m_i[r], mx[r]);
      alpha[r] = __expf(m_i[r] - mnew);
      m_i[r] = mnew;
      rs[r] = 0.0f;
    }
#pragma unroll
    for (int j = 0; j < 4; ++j)
#pragma unroll
      for (int r = 0; r < 8; ++r) {
        p[j][r] = __expf(p[j][r] - m_i[r]);
        rs[r] += p[j][r];
      }
#pragma unroll
    for (int r = 0; r < 8; ++r) {
      rs[r] += __shfl_xor(rs[r], 8, 32);
      rs[r] += __shfl_xor(rs[r], 4, 32);
      rs[r] += __shfl_xor(rs[r], 2, 32);
      rs[r] += __shfl_xor(rs[r], 1, 32);
      l_i[r] = l_i[r] * alpha[r] + rs[r];
    }
#pragma unroll
    for (int n = 0; n < 4; ++n)
#pragma unroll
      for (int r = 0; r < 8; ++r) O[n][r] *= alpha[r];

    // ---- D-layout P -> A-layout via wave-private LDS ----
#pragma unroll
    for (int j = 0; j < 4; ++j)
#pragma unroll
      for (int r = 0; r < 8; ++r)
        pw[(r + 8 * half) * 72 + j * 16 + l16] = (__bf16)p[j][r];
    asm volatile("s_wait_dscnt 0" ::: "memory");

    // ---- O += P @ V (V-frags straight from transposed global buffer) ----
#pragma unroll
    for (int d2 = 0; d2 < 2; ++d2) {
      v16bf pf = load_afrag(pw, 72, 0, d2 * 32, lane);
#pragma unroll
      for (int n = 0; n < 4; ++n) {
        v16bf vf = load_bfrag(vtb, SEQ, n * 16, k0 + d2 * 32, lane);
        O[n] = wmma_bf16(pf, vf, O[n]);
      }
    }
  }

  // ---- epilogue: normalize, store fp32 ctx in [B,S,H*Dh] layout ----
  float invl[8];
#pragma unroll
  for (int r = 0; r < 8; ++r) invl[r] = 1.0f / l_i[r];
#pragma unroll
  for (int n = 0; n < 4; ++n)
#pragma unroll
    for (int r = 0; r < 8; ++r) {
      int s  = q0 + w * 16 + r + 8 * half;
      int dh = n * 16 + l16;
      out[(size_t)(b * SEQ + s) * D_MODEL + h * DH + dh] = O[n][r] * invl[r];
    }
}

// ---------------------------------------------------------------------------
extern "C" void kernel_launch(void* const* d_in, const int* in_sizes, int n_in,
                              void* d_out, int out_size, void* d_ws, size_t ws_size,
                              hipStream_t stream) {
  (void)in_sizes; (void)n_in; (void)out_size; (void)ws_size;
  const float* x = (const float*)d_in[0];

  const size_t pstride = (size_t)MTOT * D_MODEL;          // elems per projection
  const size_t wstride = (size_t)D_MODEL * D_MODEL;       // elems per weight

  __bf16* proj = (__bf16*)d_ws;                           // 5 x [4096x768] bf16
  __bf16* xbf  = proj + 5 * pstride;                      // [4096x768] bf16
  __bf16* wbf  = xbf + pstride;                           // 5 x [768x768] bf16
  __bf16* vt   = proj + 4 * pstride;                      // V transposed [b][h][dh][s]

  // fp32 -> bf16 conversions
  cvt_f32_bf16_kernel<<<(int)(pstride / 2048), 256, 0, stream>>>(x, xbf, (int)pstride);
  for (int z = 0; z < 5; ++z)
    cvt_f32_bf16_kernel<<<(int)(wstride / 2048), 256, 0, stream>>>(
        (const float*)d_in[1 + z], wbf + z * wstride, (int)wstride);

  // 5 projection GEMMs (z selects weight; z==4 stores transposed V)
  dim3 ggrid(MTOT / BM, D_MODEL / BN, 5);
  proj_gemm_kernel<<<ggrid, 128, 0, stream>>>(xbf, wbf, proj, vt);

  // attention
  dim3 agrid(SEQ / 64, NH, NBATCH);
  attn_kernel<<<agrid, 128, 0, stream>>>(proj + 0 * pstride, proj + 1 * pstride,
                                         proj + 2 * pstride, proj + 3 * pstride,
                                         vt, (float*)d_out);
}
